// LSTM_61684320305574
// MI455X (gfx1250) — matrix operand order repaired and checked by reading, not compiled, not run on hardware
//
#include <hip/hip_runtime.h>
#include <hip/hip_bf16.h>
#include <math.h>

// Problem dims
#define BB 64
#define CC 256
#define TT 1536
#define HH 512
#define GG 2048                 // 4*H
#define M0 (BB*TT)              // 98304 rows for the batched input GEMM

typedef __attribute__((ext_vector_type(16))) __bf16 v16bf;
typedef __attribute__((ext_vector_type(8)))  float  v8f;
typedef __attribute__((ext_vector_type(4)))  unsigned tdm_u4;
typedef __attribute__((ext_vector_type(8)))  int      tdm_i8;
typedef __attribute__((ext_vector_type(4)))  int      tdm_i4;

union FragA { v16bf v; unsigned u[8]; };
union FragC { v8f   v; float    f[8]; };

// K position of the low element of VGPR v for the 16-bit 16x32 A layout
// (lanes 0-15: VGPR0..3 -> K=0..7, VGPR4..7 -> K=16..23; +8 for lanes 16-31)
__device__ __forceinline__ int a_kbase(int v, int half) {
  return (v < 4 ? 2 * v : 16 + 2 * (v - 4)) + 8 * half;
}

// ---------------------------------------------------------------------------
// Prep kernels
// ---------------------------------------------------------------------------
__global__ void f32_to_bf16(const float* __restrict__ src, __bf16* __restrict__ dst, int n) {
  int i = blockIdx.x * blockDim.x + threadIdx.x;
  if (i < n) dst[i] = (__bf16)src[i];
}

__global__ void bias_sum(const float* __restrict__ a, const float* __restrict__ b,
                         float* __restrict__ dst, int n) {
  int i = blockIdx.x * blockDim.x + threadIdx.x;
  if (i < n) dst[i] = a[i] + b[i];
}

// x[B][C][T] (f32) -> A0[b*T+t][C] (bf16), 32x32 LDS tile transpose
__global__ __launch_bounds__(256)
void transpose_x(const float* __restrict__ x, __bf16* __restrict__ A0) {
  __shared__ float tile[32][33];
  const int b  = blockIdx.z;
  const int t0 = blockIdx.x * 32;
  const int c0 = blockIdx.y * 32;
  const int tx = threadIdx.x & 31;
  const int ty = threadIdx.x >> 5;   // 0..7
#pragma unroll
  for (int l = 0; l < 4; ++l) {
    int c = ty + 8 * l;
    tile[c][tx] = x[((size_t)b * CC + (c0 + c)) * TT + (t0 + tx)];
  }
  __syncthreads();
#pragma unroll
  for (int l = 0; l < 4; ++l) {
    int tr = ty + 8 * l;  // t within tile
    A0[((size_t)b * TT + (t0 + tr)) * CC + (c0 + tx)] = (__bf16)tile[tx][tr];
  }
}

// ---------------------------------------------------------------------------
// Batched input-gate GEMM: gx[t*B+b][4H] = A[b*T+t][K] * W[4H][K]^T + bias
// 256 threads = 8 waves as 4(M) x 2(N); each wave owns a 16x64 strip:
// one A fragment reused across 4 B fragments -> 4 WMMAs per K-step.
// K is a template parameter so all strides are compile-time.
// ---------------------------------------------------------------------------
#define GEMM_RS 20   // padded LDS row stride in dwords (16 data + 4 pad)

template <int K>
__global__ __launch_bounds__(256)
void gates_gemm(const __bf16* __restrict__ A, const __bf16* __restrict__ W,
                const float* __restrict__ bias, float* __restrict__ gx) {
  __shared__ alignas(16) unsigned AsU[64 * GEMM_RS];    // 64 rows x 32 bf16
  __shared__ alignas(16) unsigned BsU[128 * GEMM_RS];   // 128 rows x 32 bf16
  const int tid  = threadIdx.x;
  const int lane = tid & 31;
  const int wave = tid >> 5;          // 0..7
  const int half = lane >> 4;
  const int l16  = lane & 15;
  const int wm   = wave >> 1;         // 0..3 (M wave)
  const int wn   = wave & 1;          // 0..1 (N wave)
  const int m_base = blockIdx.y * 64;
  const int n_base = blockIdx.x * 128;

  FragC acc[4];
#pragma unroll
  for (int j = 0; j < 4; ++j)
#pragma unroll
    for (int r = 0; r < 8; ++r) acc[j].f[r] = 0.0f;

  const int arow = tid >> 2, ac4 = tid & 3;             // A fill: 1 uint4/thread

  for (int k0 = 0; k0 < K; k0 += 32) {
    __syncthreads();
    *reinterpret_cast<uint4*>(&AsU[arow * GEMM_RS + ac4 * 4]) =
        *reinterpret_cast<const uint4*>(&A[(size_t)(m_base + arow) * K + k0 + ac4 * 8]);
#pragma unroll
    for (int q = 0; q < 2; ++q) {                        // B fill: 2 uint4/thread
      int qq = tid + q * 256;
      int row = qq >> 2, c4 = qq & 3;
      *reinterpret_cast<uint4*>(&BsU[row * GEMM_RS + c4 * 4]) =
          *reinterpret_cast<const uint4*>(&W[(size_t)(n_base + row) * K + k0 + c4 * 8]);
    }
    if (k0 + 32 < K) {                                   // prefetch next K tile
      __builtin_prefetch(&A[(size_t)(m_base + arow) * K + k0 + 32], 0, 0);
      __builtin_prefetch(&W[(size_t)(n_base + arow) * K + k0 + 32], 0, 0);
    }
    __syncthreads();

    FragA a;
#pragma unroll
    for (int v = 0; v < 8; ++v)
      a.u[v] = AsU[(wm * 16 + l16) * GEMM_RS + v + 8 * half];
#pragma unroll
    for (int j = 0; j < 4; ++j) {
      FragA bf;
#pragma unroll
      for (int v = 0; v < 8; ++v)
        bf.u[v] = BsU[(wn * 64 + j * 16 + l16) * GEMM_RS + v + 8 * half];
      acc[j].v = __builtin_amdgcn_wmma_f32_16x16x32_bf16(false, a.v, false, bf.v,
                                                         (short)0, acc[j].v, false, false);
    }
  }

#pragma unroll
  for (int j = 0; j < 4; ++j) {
    const int n_g = n_base + wn * 64 + j * 16 + l16;
    const float bv = bias[n_g];
#pragma unroll
    for (int r = 0; r < 8; ++r) {
      int row = m_base + wm * 16 + r + 8 * half;         // row = b*T + t
      int bb  = row / TT;
      int t   = row % TT;
      gx[((size_t)t * BB + bb) * GG + n_g] = acc[j].f[r] + bv;
    }
  }
}

// ---------------------------------------------------------------------------
// Persistent LSTM scan. 32 WGs x 512 threads = 512 waves = one wave per 16x16
// tile of the [64, 2048] per-step gate matrix. Whh slice lives in LDS,
// DMA'd in once by the Tensor Data Mover (with vectorized fallback fill).
// ---------------------------------------------------------------------------
#define SCAN_WGS     32
#define SCAN_THREADS 512
#define WHH_STRIDE   260   // dwords per LDS row: 256 data + 4 pad (bank spread)

__device__ __forceinline__ void grid_barrier(unsigned* counter, unsigned* gen) {
  __threadfence();                       // order this thread's global stores
  __syncthreads();
  if (threadIdx.x == 0) {
    unsigned g = __atomic_load_n(gen, __ATOMIC_RELAXED);
    if (atomicAdd(counter, 1u) + 1u == gridDim.x) {
      *counter = 0u;
      __threadfence();
      __atomic_store_n(gen, g + 1u, __ATOMIC_RELEASE);
    } else {
      while (__atomic_load_n(gen, __ATOMIC_ACQUIRE) == g) {
        __builtin_amdgcn_s_sleep(2);
      }
    }
  }
  __syncthreads();
}

__global__ __launch_bounds__(SCAN_THREADS)
void lstm_scan(const float* __restrict__ gx, const __bf16* __restrict__ Whh,
               __bf16* __restrict__ hstate, float* __restrict__ cstate,
               float* __restrict__ gbuf,
               __bf16* __restrict__ out_bf, float* __restrict__ out_f32,
               unsigned* barrier_mem) {
  extern __shared__ unsigned WhhU[];           // [64][WHH_STRIDE] dwords
  const int tid  = threadIdx.x;
  const int lane = tid & 31;
  const int wave = tid >> 5;                   // 0..15
  const int half = lane >> 4;
  const int l16  = lane & 15;
  const int mt   = wave >> 2;                  // 0..3   (M tile)
  const int ntl  = wave & 3;                   // 0..3   (local N tile)
  const int nt   = blockIdx.x * 4 + ntl;       // 0..127 (global N tile)
  unsigned* counter = barrier_mem;
  unsigned* gen     = barrier_mem + 16;

  const __bf16* slice = Whh + (size_t)blockIdx.x * 64 * HH;

#if defined(__gfx1250__) && __has_builtin(__builtin_amdgcn_tensor_load_to_lds)
  // Tensor Data Mover: DMA the 64x512 bf16 slice into LDS with 4-dword row
  // padding (pad_interval=256 dwords, pad_amount=4 dwords) -> 260-dword rows.
  if (wave == 0) {
    unsigned long long ga = (unsigned long long)(__SIZE_TYPE__)slice;
    unsigned lds_off = (unsigned)(__SIZE_TYPE__)(void*)WhhU;
    tdm_u4 g0;
    g0.x = 1u;                                          // count=1, user mode
    g0.y = lds_off;                                     // lds_addr
    g0.z = (unsigned)(ga & 0xffffffffu);                // global_addr[31:0]
    g0.w = (unsigned)((ga >> 32) & 0x1ffffffu) | 0x80000000u;  // addr[56:32] | type=2
    tdm_i8 g1 = {
      (int)((1u << 16) | (1u << 20) | (7u << 22) | (3u << 25)),  // 2B elems, pad on
      (int)(512u << 16),                                // tensor_dim0 = 512
      (int)(64u << 16),                                 // tensor_dim1 = 64
      (int)(512u << 16),                                // tile_dim0 = 512
      64,                                               // tile_dim1 = 64
      512,                                              // tensor_dim0_stride
      0, 0 };
    tdm_i4 gz  = {0, 0, 0, 0};
#if __has_include(<hip/amd_detail/amd_gfx1250_TDM.h>)
    // therock-10.0 headers -> upstream clang-23 6-arg builtin form
    tdm_i8 gz8 = {0, 0, 0, 0, 0, 0, 0, 0};
    __builtin_amdgcn_tensor_load_to_lds(g0, g1, gz, gz, gz8, 0);
#else
    // ROCm 7.2 clang-22 5-arg builtin form
    __builtin_amdgcn_tensor_load_to_lds(g0, g1, gz, gz, 0);
#endif
    __builtin_amdgcn_s_wait_tensorcnt(0);
  }
  __syncthreads();
#endif
  // Vectorized fill (idempotent overwrite of the TDM result; also the
  // portable fallback). Pads are never read, so LDS ends up correct.
  for (int idx = tid; idx < 64 * 64; idx += SCAN_THREADS) {
    int row = idx >> 6, c4 = idx & 63;
    *reinterpret_cast<uint4*>(&WhhU[row * WHH_STRIDE + c4 * 4]) =
        *reinterpret_cast<const uint4*>(&slice[(size_t)row * HH + c4 * 8]);
  }
  __syncthreads();

  const int n_g     = nt * 16 + l16;
  const int m_a     = mt * 16 + l16;
  const int n_local = ntl * 16 + l16;

  for (int t = 0; t < TT; ++t) {
    // ---- Phase 1: gate tile = gx[t] + h_prev @ Whh^T (bf16 WMMA, f32 acc)
    FragC acc;
#pragma unroll
    for (int r = 0; r < 8; ++r) {
      int m = mt * 16 + r + 8 * half;
      acc.f[r] = gx[((size_t)t * BB + m) * GG + n_g];
    }
#pragma unroll 4
    for (int kk = 0; kk < 16; ++kk) {
      FragA a, bf;
#pragma unroll
      for (int v = 0; v < 8; ++v) {
        a.u[v]  = *reinterpret_cast<const unsigned*>(
                      &hstate[m_a * HH + kk * 32 + 16 * half + 2 * v]);
        bf.u[v] = WhhU[n_local * WHH_STRIDE + kk * 16 + v + 8 * half];
      }
      acc.v = __builtin_amdgcn_wmma_f32_16x16x32_bf16(false, a.v, false, bf.v,
                                                      (short)0, acc.v, false, false);
    }
#pragma unroll
    for (int r = 0; r < 8; ++r) {
      int m = mt * 16 + r + 8 * half;
      gbuf[m * GG + n_g] = acc.f[r];
    }
    __builtin_amdgcn_s_cluster_barrier();   // HW sync if clustered (NOP otherwise)
    grid_barrier(counter, gen);

    // ---- Phase 2: elementwise LSTM cell (i,f,g,o gate order)
    for (int e = blockIdx.x * SCAN_THREADS + tid; e < BB * HH;
         e += SCAN_WGS * SCAN_THREADS) {
      int m = e >> 9, j = e & 511;
      float gi = gbuf[m * GG + j];
      float gf = gbuf[m * GG + 512 + j];
      float gg = gbuf[m * GG + 1024 + j];
      float go = gbuf[m * GG + 1536 + j];
      gi = 1.0f / (1.0f + __expf(-gi));
      gf = 1.0f / (1.0f + __expf(-gf));
      go = 1.0f / (1.0f + __expf(-go));
      gg = tanhf(gg);
      float c = gf * cstate[e] + gi * gg;
      cstate[e] = c;
      float h = go * tanhf(c);
      hstate[e] = (__bf16)h;
      size_t oidx = ((size_t)m * TT + t) * HH + j;   // [B][T][H]
      if (out_bf) out_bf[oidx] = (__bf16)h;
      else        out_f32[oidx] = h;
    }
    __builtin_amdgcn_s_cluster_barrier();
    grid_barrier(counter, gen);
  }
}

// ---------------------------------------------------------------------------
// Host orchestration
// ---------------------------------------------------------------------------
extern "C" void kernel_launch(void* const* d_in, const int* in_sizes, int n_in,
                              void* d_out, int out_size, void* d_ws, size_t ws_size,
                              hipStream_t stream) {
  const float* x    = (const float*)d_in[0];
  const float* Wih0 = (const float*)d_in[1];
  const float* Whh0 = (const float*)d_in[2];
  const float* bih0 = (const float*)d_in[3];
  const float* bhh0 = (const float*)d_in[4];
  const float* Wih1 = (const float*)d_in[5];
  const float* Whh1 = (const float*)d_in[6];
  const float* bih1 = (const float*)d_in[7];
  const float* bhh1 = (const float*)d_in[8];
  float* out = (float*)d_out;

  char* ws = (char*)d_ws;
  size_t off = 0;
  auto walloc = [&](size_t bytes) -> void* {
    void* p = ws + off;
    off = (off + bytes + 255) & ~(size_t)255;
    return p;
  };
  float*    gx    = (float*)  walloc((size_t)TT * BB * GG * 4);  // [T][B][4H]
  __bf16*   A0    = (__bf16*) walloc((size_t)M0 * CC * 2);       // x transposed, bf16
  __bf16*   H0    = (__bf16*) walloc((size_t)M0 * HH * 2);       // layer-0 h output
  __bf16*   W0i   = (__bf16*) walloc((size_t)GG * CC * 2);
  __bf16*   W0h   = (__bf16*) walloc((size_t)GG * HH * 2);
  __bf16*   W1i   = (__bf16*) walloc((size_t)GG * HH * 2);
  __bf16*   W1h   = (__bf16*) walloc((size_t)GG * HH * 2);
  float*    bias0 = (float*)  walloc((size_t)GG * 4);
  float*    bias1 = (float*)  walloc((size_t)GG * 4);
  __bf16*   hst   = (__bf16*) walloc((size_t)BB * HH * 2);
  float*    cst   = (float*)  walloc((size_t)BB * HH * 4);
  float*    gbuf  = (float*)  walloc((size_t)BB * GG * 4);
  unsigned* barm  = (unsigned*)walloc(256);

  (void)hipMemsetAsync(barm, 0, 256, stream);

  f32_to_bf16<<<(GG * CC + 255) / 256, 256, 0, stream>>>(Wih0, W0i, GG * CC);
  f32_to_bf16<<<(GG * HH + 255) / 256, 256, 0, stream>>>(Whh0, W0h, GG * HH);
  f32_to_bf16<<<(GG * HH + 255) / 256, 256, 0, stream>>>(Wih1, W1i, GG * HH);
  f32_to_bf16<<<(GG * HH + 255) / 256, 256, 0, stream>>>(Whh1, W1h, GG * HH);
  bias_sum<<<(GG + 255) / 256, 256, 0, stream>>>(bih0, bhh0, bias0, GG);
  bias_sum<<<(GG + 255) / 256, 256, 0, stream>>>(bih1, bhh1, bias1, GG);
  transpose_x<<<dim3(TT / 32, CC / 32, BB), 256, 0, stream>>>(x, A0);

  const size_t scan_lds = (size_t)64 * WHH_STRIDE * 4;  // ~65 KB

  // Layer 0
  gates_gemm<CC><<<dim3(GG / 128, M0 / 64), 256, 0, stream>>>(A0, W0i, bias0, gx);
  (void)hipMemsetAsync(hst, 0, (size_t)BB * HH * 2, stream);
  (void)hipMemsetAsync(cst, 0, (size_t)BB * HH * 4, stream);
  lstm_scan<<<SCAN_WGS, SCAN_THREADS, scan_lds, stream>>>(gx, W0h, hst, cst, gbuf,
                                                          H0, nullptr, barm);
  // Layer 1
  gates_gemm<HH><<<dim3(GG / 128, M0 / 64), 256, 0, stream>>>(H0, W1i, bias1, gx);
  (void)hipMemsetAsync(hst, 0, (size_t)BB * HH * 2, stream);
  (void)hipMemsetAsync(cst, 0, (size_t)BB * HH * 4, stream);
  lstm_scan<<<SCAN_WGS, SCAN_THREADS, scan_lds, stream>>>(gx, W1h, hst, cst, gbuf,
                                                          nullptr, out, barm);
}
